// MultiHeadAttention_37228776521777
// MI455X (gfx1250) — matrix operand order repaired and checked by reading, not compiled, not run on hardware
//
#include <hip/hip_runtime.h>

// ---------------------------------------------------------------------------
// MI455X (gfx1250) multi-head attention forward, bf16 WMMA pipeline.
// wave32; V_WMMA_F32_16X16X32_BF16 + GLOBAL_LOAD_ASYNC_TO_LDS_B128 staging.
// ---------------------------------------------------------------------------

typedef __attribute__((ext_vector_type(16))) __bf16 v16bf;
typedef __attribute__((ext_vector_type(8)))  __bf16 v8bf;
typedef __attribute__((ext_vector_type(8)))  float  v8f;

#define SEQ 4096
#define DM  1024
#define NH  16
#define DH  64

__device__ inline __bf16 f32_to_bf16(float f) {
  unsigned int u = __builtin_bit_cast(unsigned int, f);
  unsigned int r = u + 0x7FFFu + ((u >> 16) & 1u);   // round to nearest even
  unsigned short h = (unsigned short)(r >> 16);
  return __builtin_bit_cast(__bf16, h);
}

// ds_swizzle xor within 32-lane group (and_mask=0x1f); xor<=8 stays in 16-lane half
template <int MASK>
__device__ inline float swzf(float x) {
  int i = __builtin_amdgcn_ds_swizzle(__builtin_bit_cast(int, x), (MASK << 10) | 0x1F);
  return __builtin_bit_cast(float, i);
}
__device__ inline float rowmax16(float x) {
  x = fmaxf(x, swzf<1>(x)); x = fmaxf(x, swzf<2>(x));
  x = fmaxf(x, swzf<4>(x)); x = fmaxf(x, swzf<8>(x));
  return x;
}
__device__ inline float rowsum16(float x) {
  x += swzf<1>(x); x += swzf<2>(x); x += swzf<4>(x); x += swzf<8>(x);
  return x;
}

// A-operand tile loader: 16x32 bf16 (M x K), row-major source with leading dim ld.
// ISA layout: lane m=lane&15 holds row m; K-chunks {kb..kb+7, kb+16..kb+23}, kb=(lane>>4)*8
__device__ inline v16bf load_a_tile(const __bf16* __restrict__ base, int ld, int lane) {
  int m  = lane & 15;
  int kb = (lane >> 4) << 3;
  const __bf16* p = base + m * ld + kb;
  v8bf a0 = *(const v8bf*)(p);
  v8bf a1 = *(const v8bf*)(p + 16);
  v16bf r{};
#pragma unroll
  for (int i = 0; i < 8; ++i) { r[i] = a0[i]; r[i + 8] = a1[i]; }
  return r;
}

__device__ inline v8f wmma_bf16(v16bf a, v16bf b, v8f c) {
  return __builtin_amdgcn_wmma_f32_16x16x32_bf16(false, a, false, b, (short)0, c,
                                                 false, false);
}

// Async copy of 16 bytes/lane global -> LDS (VDST VGPR = LDS byte offset; generic
// pointers to __shared__ carry the LDS offset in their low 32 bits per the
// gfx1250 aperture rules). Tracked by ASYNCcnt.
__device__ inline void async_g2l_b128(unsigned lds_off, const void* gaddr) {
  asm volatile("global_load_async_to_lds_b128 %0, %1, off"
               :: "v"(lds_off), "v"((unsigned long long)(uintptr_t)gaddr)
               : "memory");
}

// ---------------------------------------------------------------------------
// fp32 -> bf16 conversion
// ---------------------------------------------------------------------------
__global__ void cvt_f32_bf16_k(const float* __restrict__ in, __bf16* __restrict__ out, int n) {
  int i = blockIdx.x * blockDim.x + threadIdx.x;
  int stride = gridDim.x * blockDim.x;
  for (; i < n; i += stride) out[i] = f32_to_bf16(in[i]);
}

// ---------------------------------------------------------------------------
// Block-cooperative bf16 GEMM: C(MxN) = A(MxK) * B(KxN).
// Block = 4 waves -> 128x64 C tile; each wave owns 32x64 (two stacked 16-row
// A tiles, 8 accumulators) and shares a double-buffered async-staged B tile.
// mode 0: C bf16 row-major     (Q, V)
// mode 1: C bf16 transposed    (Kt[d][seq])
// mode 2: C f32 + bias         (output projection)
// ---------------------------------------------------------------------------
#define BPAD 72   // 32x64 bf16 tile, rows padded to 72 elems (144B, 16B-aligned)

__global__ void __launch_bounds__(128)
gemm_bf16_block_k(const __bf16* __restrict__ A, const __bf16* __restrict__ B,
                  __bf16* __restrict__ Cb, float* __restrict__ Cf,
                  const float* __restrict__ bias,
                  int M, int N, int K, int mode) {
  __shared__ __align__(64) __bf16 bsh[2][32][BPAD];
  int tid  = threadIdx.x;
  int wib  = tid >> 5;
  int lane = tid & 31;
  int nbn  = N >> 6;
  int m0 = (blockIdx.x / nbn) * 128 + wib * 32;
  int n0 = (blockIdx.x % nbn) * 64;

  const int T = K >> 5;    // number of 32-deep K tiles

  // Issue the whole 4 KB B tile for K-step t into buffer buf:
  // 128 threads x 2 chunks x 16 B. Each wave contributes 2 async instructions.
  auto issue_tile = [&](int t, int buf) {
#pragma unroll
    for (int i = 0; i < 2; ++i) {
      int o   = (i * 128 + tid) * 16;   // byte offset within packed 32x128B tile
      int row = o >> 7;                 // 128 B per logical row
      int cb  = o & 127;
      unsigned lds = (unsigned)(uintptr_t)&bsh[buf][0][0] + row * (BPAD * 2) + cb;
      const char* g = (const char*)B + ((size_t)(t * 32 + row) * N + n0) * 2 + cb;
      async_g2l_b128(lds, g);
    }
  };

  v8f acc[2][4] = {};
  auto compute = [&](int t) {
    v16bf a0 = load_a_tile(A + (size_t)m0 * K + t * 32,        K, lane);
    v16bf a1 = load_a_tile(A + (size_t)(m0 + 16) * K + t * 32, K, lane);
    const __bf16* bcol = &bsh[t & 1][(lane >> 4) << 3][lane & 15];   // lane&15 = N col; K rows {kb..kb+7, kb+16..kb+23}, kb=(lane>>4)*8
#pragma unroll
    for (int j = 0; j < 4; ++j) {
      v16bf b;
#pragma unroll
      for (int i = 0; i < 8; ++i) { b[i] = bcol[i * BPAD + j * 16]; b[i + 8] = bcol[(16 + i) * BPAD + j * 16]; }
      acc[0][j] = wmma_bf16(a0, b, acc[0][j]);
      acc[1][j] = wmma_bf16(a1, b, acc[1][j]);
    }
  };

  issue_tile(0, 0);
#pragma unroll 2
  for (int t = 0; t < T - 1; ++t) {
    issue_tile(t + 1, (t + 1) & 1);
    asm volatile("s_wait_asynccnt 0x2" ::: "memory");  // tile t complete (in-order)
    __syncthreads();                                   // all waves' chunks visible
    compute(t);
    __syncthreads();                                   // reads done before buf reuse
  }
  asm volatile("s_wait_asynccnt 0x0" ::: "memory");
  __syncthreads();
  compute(T - 1);

  int hi = lane >> 4, lo = lane & 15;
#pragma unroll
  for (int g = 0; g < 2; ++g) {
#pragma unroll
    for (int j = 0; j < 4; ++j) {
#pragma unroll
      for (int e = 0; e < 8; ++e) {
        int row = m0 + g * 16 + e + hi * 8;
        int col = n0 + j * 16 + lo;
        if (mode == 2)      Cf[row * N + col] = acc[g][j][e] + bias[col];
        else if (mode == 1) Cb[col * M + row] = f32_to_bf16(acc[g][j][e]);
        else                Cb[row * N + col] = f32_to_bf16(acc[g][j][e]);
      }
    }
  }
}

// ---------------------------------------------------------------------------
// Causal flash attention: one wave = 16 queries of one head.
// Q:[SEQ][DM] bf16, Kt:[DM][SEQ] bf16 (head-major rows), V:[SEQ][DM] bf16.
// ---------------------------------------------------------------------------
__global__ void __launch_bounds__(128)
attn_k(const __bf16* __restrict__ Q, const __bf16* __restrict__ Kt,
       const __bf16* __restrict__ V, __bf16* __restrict__ Ctx) {
  __shared__ __align__(64) __bf16 plds[4][16][32];   // per-wave P staging tile (1 KB)
  int wib  = threadIdx.x >> 5;
  int wave = blockIdx.x * 4 + wib;
  int lane = threadIdx.x & 31;
  int hi = lane >> 4, lo = lane & 15;
  int head = wave & (NH - 1);
  int q0   = (wave >> 4) << 4;

  const __bf16* qbase = Q + q0 * DM + head * DH;
  v16bf aq0 = load_a_tile(qbase,      DM, lane);   // d = 0..31
  v16bf aq1 = load_a_tile(qbase + 32, DM, lane);   // d = 32..63

  v8f acc[4] = {};
  float mrow[8], srow[8];
#pragma unroll
  for (int e = 0; e < 8; ++e) { mrow[e] = -1e30f; srow[e] = 0.f; }

  const float scale = 0.125f;                  // 1/sqrt(64)
  int nkb = (q0 >> 5) + 1;                     // 32-key blocks covering keys <= q0+15
  const __bf16* ktrow0 = Kt + (head * DH + hi * 8) * SEQ + lo;       // key = lo; d = {8hi..8hi+7, 16+8hi..}
  const __bf16* ktrow1 = Kt + (head * DH + 32 + hi * 8) * SEQ + lo;  // key = lo; d = 32 + same

  for (int kb = 0; kb < nkb; ++kb) {
    int key0 = kb << 5;
    // ---- S = Q * K^T (16 queries x 32 keys) ----
    v8f s[2] = {};
#pragma unroll
    for (int t = 0; t < 2; ++t) {
      v16bf b0, b1;
#pragma unroll
      for (int i = 0; i < 8; ++i) {
        b0[i] = ktrow0[i * SEQ + key0 + t * 16]; b0[i + 8] = ktrow0[(16 + i) * SEQ + key0 + t * 16];
        b1[i] = ktrow1[i * SEQ + key0 + t * 16]; b1[i + 8] = ktrow1[(16 + i) * SEQ + key0 + t * 16];
      }
      s[t] = wmma_bf16(aq0, b0, s[t]);
      s[t] = wmma_bf16(aq1, b1, s[t]);
    }
    // ---- online softmax update (rows live in 16-lane halves) ----
    float p0[8], p1[8];
#pragma unroll
    for (int e = 0; e < 8; ++e) {
      int qi = q0 + e + hi * 8;
      float s0 = (key0 + lo      <= qi) ? s[0][e] * scale : -1e30f;
      float s1 = (key0 + 16 + lo <= qi) ? s[1][e] * scale : -1e30f;
      float rmax = rowmax16(fmaxf(s0, s1));
      float mnew = fmaxf(mrow[e], rmax);
      float corr = __expf(mrow[e] - mnew);
      float e0 = __expf(s0 - mnew);
      float e1 = __expf(s1 - mnew);
      float rs = rowsum16(e0 + e1);
      srow[e] = srow[e] * corr + rs;
      mrow[e] = mnew;
#pragma unroll
      for (int j = 0; j < 4; ++j) acc[j][e] *= corr;
      p0[e] = e0; p1[e] = e1;
    }
    // ---- C-layout -> A-layout via LDS (same-wave DS ops are in-order) ----
#pragma unroll
    for (int e = 0; e < 8; ++e) {
      plds[wib][e + hi * 8][lo]      = f32_to_bf16(p0[e]);
      plds[wib][e + hi * 8][16 + lo] = f32_to_bf16(p1[e]);
    }
    asm volatile("s_wait_dscnt 0x0" ::: "memory");
    v16bf pa{};
    {
      int kbA = hi << 3;
#pragma unroll
      for (int i = 0; i < 8; ++i) {
        pa[i]     = plds[wib][lo][kbA + i];
        pa[i + 8] = plds[wib][lo][kbA + 16 + i];
      }
    }
    asm volatile("" ::: "memory");
    // ---- acc += P * V (contract over 32 keys; lo = d column for B, elements = keys) ----
    const __bf16* vrow = V + (key0 + hi * 8) * DM + head * DH + lo;
#pragma unroll
    for (int j = 0; j < 4; ++j) {
      v16bf bv;
#pragma unroll
      for (int i = 0; i < 8; ++i) { bv[i] = vrow[i * DM + j * 16]; bv[i + 8] = vrow[(16 + i) * DM + j * 16]; }
      acc[j] = wmma_bf16(pa, bv, acc[j]);
    }
  }
  // ---- normalize and store ctx ----
#pragma unroll
  for (int e = 0; e < 8; ++e) {
    float inv = 1.0f / srow[e];
    int row = q0 + e + hi * 8;
#pragma unroll
    for (int j = 0; j < 4; ++j)
      Ctx[row * DM + head * DH + j * 16 + lo] = f32_to_bf16(acc[j][e] * inv);
  }
}

// ---------------------------------------------------------------------------
extern "C" void kernel_launch(void* const* d_in, const int* in_sizes, int n_in,
                              void* d_out, int out_size, void* d_ws, size_t ws_size,
                              hipStream_t stream) {
  const float* x  = (const float*)d_in[0];
  const float* Wq = (const float*)d_in[1];
  const float* Wk = (const float*)d_in[2];
  const float* Wv = (const float*)d_in[3];
  const float* Wo = (const float*)d_in[4];
  const float* bo = (const float*)d_in[5];

  char* ws = (char*)d_ws;
  const size_t MB = (size_t)1 << 20;
  __bf16* xb  = (__bf16*)(ws + 0 * MB);   // 8 MB  : x bf16
  __bf16* wqb = (__bf16*)(ws + 8 * MB);   // 2 MB
  __bf16* wkb = (__bf16*)(ws + 10 * MB);  // 2 MB
  __bf16* wvb = (__bf16*)(ws + 12 * MB);  // 2 MB
  __bf16* wob = (__bf16*)(ws + 14 * MB);  // 2 MB
  __bf16* Qb  = (__bf16*)(ws + 16 * MB);  // 8 MB  : Q  [SEQ][DM]
  __bf16* Ktb = (__bf16*)(ws + 24 * MB);  // 8 MB  : Kt [DM][SEQ]
  __bf16* Vb  = (__bf16*)(ws + 32 * MB);  // 8 MB  : V  [SEQ][DM]
  __bf16* Cx  = (__bf16*)(ws + 40 * MB);  // 8 MB  : ctx[SEQ][DM]

  cvt_f32_bf16_k<<<1024, 256, 0, stream>>>(x,  xb,  SEQ * DM);
  cvt_f32_bf16_k<<<512,  256, 0, stream>>>(Wq, wqb, DM * DM);
  cvt_f32_bf16_k<<<512,  256, 0, stream>>>(Wk, wkb, DM * DM);
  cvt_f32_bf16_k<<<512,  256, 0, stream>>>(Wv, wvb, DM * DM);
  cvt_f32_bf16_k<<<512,  256, 0, stream>>>(Wo, wob, DM * DM);

  // (SEQ/128)*(DM/64) = 512 blocks of 4 waves (128x64 C tile per block)
  gemm_bf16_block_k<<<512, 128, 0, stream>>>(xb, wqb, Qb,  nullptr, nullptr, SEQ, DM, DM, 0);
  gemm_bf16_block_k<<<512, 128, 0, stream>>>(xb, wkb, Ktb, nullptr, nullptr, SEQ, DM, DM, 1);
  gemm_bf16_block_k<<<512, 128, 0, stream>>>(xb, wvb, Vb,  nullptr, nullptr, SEQ, DM, DM, 0);

  // 256 query tiles * 16 heads = 4096 waves -> 1024 blocks
  attn_k<<<1024, 128, 0, stream>>>(Qb, Ktb, Vb, Cx);

  gemm_bf16_block_k<<<512, 128, 0, stream>>>(Cx, wob, nullptr, (float*)d_out, bo,
                                             SEQ, DM, DM, 2);
}
